// Net_40037685133503
// MI455X (gfx1250) — matrix operand order repaired and checked by reading, not compiled, not run on hardware
//
#include <hip/hip_runtime.h>

// ---------- types ----------
typedef __attribute__((ext_vector_type(16))) __bf16          v16bf;
typedef __attribute__((ext_vector_type(16))) unsigned short  v16u;
typedef __attribute__((ext_vector_type(8)))  float           v8f;
typedef int v4i __attribute__((vector_size(16)));            // matches builtin param

union V16 { v16u u; v16bf b; };

__device__ __forceinline__ unsigned short f2bf(float f) {
    unsigned u = __float_as_uint(f);
    unsigned r = (u + 0x7FFFu + ((u >> 16) & 1u)) >> 16;
    return (unsigned short)r;
}
__device__ __forceinline__ float bf2f(unsigned short u) {
    return __uint_as_float(((unsigned)u) << 16);
}
// 16-bit A fragment: element e of lane-half h maps to local K:
//   k = e + 16*h
__device__ __forceinline__ int kA_local(int e, int h) {
    return e + (h << 4);
}

// ---------- gfx1250 async global->LDS staging (guarded) ----------
#if defined(__has_builtin)
#if __has_builtin(__builtin_amdgcn_global_load_async_to_lds_b128)
#define USE_ASYNC_LDS 1
#endif
#endif
#ifndef USE_ASYNC_LDS
#define USE_ASYNC_LDS 0
#endif

typedef __attribute__((address_space(1))) v4i* gv4i_ptr;   // global int4*
typedef __attribute__((address_space(3))) v4i* lv4i_ptr;   // LDS int4*

__device__ __forceinline__ void async_copy16(const void* g, void* l) {
#if USE_ASYNC_LDS
    __builtin_amdgcn_global_load_async_to_lds_b128((gv4i_ptr)g, (lv4i_ptr)l, 0, 0);
#endif
}
__device__ __forceinline__ void wait_async() {
#if USE_ASYNC_LDS
#if __has_builtin(__builtin_amdgcn_s_wait_asynccnt)
    __builtin_amdgcn_s_wait_asynccnt(0);
#else
    asm volatile("s_wait_asynccnt 0x0" ::: "memory");
#endif
#endif
}

// =====================================================================
// Prep: pre-shuffle conv weights so every A fragment is 32 contiguous B.
// =====================================================================
__global__ __launch_bounds__(256) void k_prep(const float* __restrict__ w2,
                                              const float* __restrict__ w3,
                                              unsigned short* __restrict__ wp2,
                                              unsigned short* __restrict__ wp3) {
    int tid = blockIdx.x * 256 + threadIdx.x;
    for (int idx = tid; idx < 9 * 64 * 32; idx += 32 * 256) {
        int rs = idx >> 11, rem = idx & 2047;
        int m = rem >> 5, q = rem & 31;
        int h = q >> 4, e = q & 15;
        wp2[idx] = f2bf(w2[m * 288 + kA_local(e, h) * 9 + rs]);
    }
    for (int idx = tid; idx < 9 * 64 * 64; idx += 32 * 256) {
        int rs = idx >> 12, rem = idx & 4095;
        int m = rem >> 6, q = rem & 63;
        int ks = q >> 5, h = (q >> 4) & 1, e = q & 15;
        wp3[idx] = f2bf(w3[m * 576 + (ks * 32 + kA_local(e, h)) * 9 + rs]);
    }
}

// =====================================================================
// Kernel 1: conv1 (1->32) + IN + ReLU -> h1 in [p][32] bf16.
// Input staged as zero-padded 30x30 plane -> branch-free 3x3 window.
// =====================================================================
__global__ __launch_bounds__(256) void k_conv1(const float* __restrict__ x,
                                               const float* __restrict__ w1,
                                               const float* __restrict__ b1,
                                               unsigned short* __restrict__ h1t) {
    extern __shared__ char smem[];
    float* sInP = (float*)smem;          // 30*30 padded
    float* sW   = sInP + 900;            // 288
    float* sB   = sW + 288;              // 32
    float* sOut = sB + 32;               // 32*784  [c][p]
    float* sMu  = sOut + 32 * 784;       // 32
    float* sRsq = sMu + 32;              // 32
    const int n = blockIdx.x;
    const float* xin = x + (size_t)n * 784;
    for (int i = threadIdx.x; i < 900; i += 256) sInP[i] = 0.f;
    for (int i = threadIdx.x; i < 288; i += 256) sW[i] = w1[i];
    if (threadIdx.x < 32) sB[threadIdx.x] = b1[threadIdx.x];
    __syncthreads();
    for (int i = threadIdx.x; i < 784; i += 256) {
        int yy = i / 28, xx = i - yy * 28;
        sInP[(yy + 1) * 30 + (xx + 1)] = xin[i];
    }
    __syncthreads();

    // per-thread output positions (one-time divisions)
    int base[4];
    #pragma unroll
    for (int j = 0; j < 4; ++j) {
        int p = threadIdx.x + j * 256;
        int yy = p / 28, xx = p - yy * 28;
        base[j] = yy * 30 + xx;          // top-left of padded window
    }
    const bool act3 = threadIdx.x < 16;  // j==3 valid only for p<784

    for (int c = 0; c < 32; ++c) {
        const float* wc = sW + c * 9;
        float w0 = wc[0], w1r = wc[1], w2r = wc[2], w3r = wc[3], w4r = wc[4],
              w5r = wc[5], w6r = wc[6], w7r = wc[7], w8r = wc[8];
        float bb = sB[c];
        #pragma unroll
        for (int j = 0; j < 4; ++j) {
            if (j == 3 && !act3) break;
            const float* bp = sInP + base[j];
            float acc = bb;
            acc += bp[0]  * w0;  acc += bp[1]  * w1r; acc += bp[2]  * w2r;
            acc += bp[30] * w3r; acc += bp[31] * w4r; acc += bp[32] * w5r;
            acc += bp[60] * w6r; acc += bp[61] * w7r; acc += bp[62] * w8r;
            sOut[c * 784 + threadIdx.x + j * 256] = acc;
        }
    }
    __syncthreads();

    const int wave = threadIdx.x >> 5, lane = threadIdx.x & 31;
    for (int c = wave; c < 32; c += 8) {
        float s = 0.f, s2 = 0.f;
        for (int p = lane; p < 784; p += 32) { float v = sOut[c * 784 + p]; s += v; s2 += v * v; }
        #pragma unroll
        for (int mk = 16; mk >= 1; mk >>= 1) { s += __shfl_xor(s, mk, 32); s2 += __shfl_xor(s2, mk, 32); }
        if (lane == 0) {
            float mu  = s  * (1.f / 784.f);
            float var = s2 * (1.f / 784.f) - mu * mu;
            sMu[c] = mu; sRsq[c] = rsqrtf(var + 1e-5f);
        }
    }
    __syncthreads();
    unsigned* dst32 = (unsigned*)(h1t + (size_t)n * 25088);
    for (int i = threadIdx.x; i < 12544; i += 256) {
        int p = i >> 4, c2 = (i & 15) << 1;
        float v0 = (sOut[c2 * 784 + p] - sMu[c2]) * sRsq[c2];
        float v1 = (sOut[(c2 + 1) * 784 + p] - sMu[c2 + 1]) * sRsq[c2 + 1];
        v0 = v0 > 0.f ? v0 : 0.f;
        v1 = v1 > 0.f ? v1 : 0.f;
        dst32[i] = (unsigned)f2bf(v0) | ((unsigned)f2bf(v1) << 16);
    }
}

// =====================================================================
// Kernel 2: conv2 (32->64) as 9 shifted GEMMs (bf16 WMMA) + IN.
// Zero guard row at p=784 makes B loads unconditional.
// =====================================================================
__global__ __launch_bounds__(256) void k_conv2(const unsigned short* __restrict__ h1t,
                                               const unsigned short* __restrict__ wp2,
                                               const float* __restrict__ b2,
                                               unsigned short* __restrict__ h2t) {
    extern __shared__ char smem[];
    unsigned short* sInT = (unsigned short*)smem;    // [785][32] bf16 (row 784 = zeros)
    float* sOut = (float*)(sInT + 785 * 32);         // [64][784] f32
    float* sMu  = sOut + 64 * 784;                   // 64
    float* sRsq = sMu + 64;                          // 64
    const int n = blockIdx.x;
    {
        const uint4* src = (const uint4*)(h1t + (size_t)n * 25088);
        uint4* dst = (uint4*)sInT;
        if (threadIdx.x < 4) dst[3136 + threadIdx.x] = make_uint4(0, 0, 0, 0);
#if USE_ASYNC_LDS
        for (int i = threadIdx.x; i < 3136; i += 256) async_copy16(src + i, dst + i);
        wait_async();
#else
        for (int i = threadIdx.x; i < 3136; i += 256) dst[i] = src[i];
#endif
    }
    __syncthreads();

    const int wave = threadIdx.x >> 5, lane = threadIdx.x & 31;
    const int half = lane >> 4, lm = lane & 15;
    const int mt = wave & 3;

    v16bf a9[9];
    #pragma unroll
    for (int rs = 0; rs < 9; ++rs) {
        V16 t; t.u = *(const v16u*)(wp2 + (rs * 64 + mt * 16 + lm) * 32 + half * 16);
        a9[rs] = t.b;
    }
    float bias[8];
    #pragma unroll
    for (int r = 0; r < 8; ++r) bias[r] = b2[mt * 16 + r + half * 8];

    const int nt0 = wave >> 2;
    int nn = nt0 * 16 + lm;
    int oy = nn / 28;                    // one-time division
    int ox = nn - oy * 28;
    for (int nt = nt0; nt < 49; nt += 2) {
        const int rowbase = oy * 28;
        v8f acc = {};
        #pragma unroll
        for (int r = 0; r < 3; ++r) {
            #pragma unroll
            for (int s = 0; s < 3; ++s) {
                int iy = oy + r - 1, ix = ox + s - 1;
                bool valid = ((unsigned)iy < 28u) & ((unsigned)ix < 28u);
                int p2 = valid ? (rowbase + (r - 1) * 28 + ix) : 784;
                V16 tb; tb.u = *(const v16u*)(sInT + p2 * 32 + half * 16);
                acc = __builtin_amdgcn_wmma_f32_16x16x32_bf16(false, a9[r * 3 + s], false, tb.b,
                                                              (short)0, acc, false, false);
            }
        }
        #pragma unroll
        for (int r = 0; r < 8; ++r)
            sOut[(mt * 16 + r + half * 8) * 784 + nn] = acc[r] + bias[r];
        nn += 32; ox += 4; oy += 1;
        if (ox >= 28) { ox -= 28; oy += 1; }
    }
    __syncthreads();

    for (int c = wave; c < 64; c += 8) {
        float s = 0.f, s2 = 0.f;
        for (int p = lane; p < 784; p += 32) { float v = sOut[c * 784 + p]; s += v; s2 += v * v; }
        #pragma unroll
        for (int mk = 16; mk >= 1; mk >>= 1) { s += __shfl_xor(s, mk, 32); s2 += __shfl_xor(s2, mk, 32); }
        if (lane == 0) {
            float mu  = s  * (1.f / 784.f);
            float var = s2 * (1.f / 784.f) - mu * mu;
            sMu[c] = mu; sRsq[c] = rsqrtf(var + 1e-5f);
        }
    }
    __syncthreads();
    unsigned* dst32 = (unsigned*)(h2t + (size_t)n * 50176);
    for (int i = threadIdx.x; i < 25088; i += 256) {
        int p = i >> 5, c2 = (i & 31) << 1;
        float v0 = (sOut[c2 * 784 + p] - sMu[c2]) * sRsq[c2];
        float v1 = (sOut[(c2 + 1) * 784 + p] - sMu[c2 + 1]) * sRsq[c2 + 1];
        dst32[i] = (unsigned)f2bf(v0) | ((unsigned)f2bf(v1) << 16);
    }
}

// =====================================================================
// Kernel 3: conv3 (64->64): 9 shifted GEMMs x 2 k-steps + ReLU + maxpool.
// All 18 A fragments hoisted; zero guard row for B.
// =====================================================================
__global__ __launch_bounds__(256) void k_conv3(const unsigned short* __restrict__ h2t,
                                               const unsigned short* __restrict__ wp3,
                                               const float* __restrict__ b3,
                                               unsigned short* __restrict__ h3) {
    extern __shared__ char smem[];
    unsigned short* sInT = (unsigned short*)smem;    // [785][64] bf16 (row 784 = zeros)
    float* sOut = (float*)(sInT + 785 * 64);         // [64][784] f32
    const int n = blockIdx.x;
    {
        const uint4* src = (const uint4*)(h2t + (size_t)n * 50176);
        uint4* dst = (uint4*)sInT;
        if (threadIdx.x < 8) dst[6272 + threadIdx.x] = make_uint4(0, 0, 0, 0);
#if USE_ASYNC_LDS
        for (int i = threadIdx.x; i < 6272; i += 256) async_copy16(src + i, dst + i);
        wait_async();
#else
        for (int i = threadIdx.x; i < 6272; i += 256) dst[i] = src[i];
#endif
    }
    __syncthreads();

    const int wave = threadIdx.x >> 5, lane = threadIdx.x & 31;
    const int half = lane >> 4, lm = lane & 15;
    const int mt = wave & 3;

    v16bf a18[18];
    #pragma unroll
    for (int rs = 0; rs < 9; ++rs) {
        #pragma unroll
        for (int ks = 0; ks < 2; ++ks) {
            V16 t; t.u = *(const v16u*)(wp3 + ((rs * 64 + mt * 16 + lm) * 2 + ks) * 32 + half * 16);
            a18[rs * 2 + ks] = t.b;
        }
    }
    float bias[8];
    #pragma unroll
    for (int r = 0; r < 8; ++r) bias[r] = b3[mt * 16 + r + half * 8];

    const int nt0 = wave >> 2;
    int nn = nt0 * 16 + lm;
    int oy = nn / 28;
    int ox = nn - oy * 28;
    for (int nt = nt0; nt < 49; nt += 2) {
        const int rowbase = oy * 28;
        v8f acc = {};
        #pragma unroll
        for (int r = 0; r < 3; ++r) {
            #pragma unroll
            for (int s = 0; s < 3; ++s) {
                const int rs = r * 3 + s;
                int iy = oy + r - 1, ix = ox + s - 1;
                bool valid = ((unsigned)iy < 28u) & ((unsigned)ix < 28u);
                int p2 = valid ? (rowbase + (r - 1) * 28 + ix) : 784;
                #pragma unroll
                for (int ks = 0; ks < 2; ++ks) {
                    V16 tb; tb.u = *(const v16u*)(sInT + p2 * 64 + ks * 32 + half * 16);
                    acc = __builtin_amdgcn_wmma_f32_16x16x32_bf16(false, a18[rs * 2 + ks], false, tb.b,
                                                                  (short)0, acc, false, false);
                }
            }
        }
        #pragma unroll
        for (int r = 0; r < 8; ++r) {
            float v = acc[r] + bias[r];
            sOut[(mt * 16 + r + half * 8) * 784 + nn] = v > 0.f ? v : 0.f;
        }
        nn += 32; ox += 4; oy += 1;
        if (ox >= 28) { ox -= 28; oy += 1; }
    }
    __syncthreads();

    unsigned* dst32 = (unsigned*)(h3 + (size_t)n * 12544);
    for (int i = threadIdx.x; i < 6272; i += 256) {
        int k = i * 2;
        int c = k / 196, p = k - c * 196;
        int py0 = p / 14, px0 = p - py0 * 14;
        int p1 = p + 1;
        int py1 = p1 / 14, px1 = p1 - py1 * 14;
        const float* b0 = sOut + c * 784 + 2 * py0 * 28 + 2 * px0;
        const float* b1 = sOut + c * 784 + 2 * py1 * 28 + 2 * px1;
        float v0 = fmaxf(fmaxf(b0[0], b0[1]), fmaxf(b0[28], b0[29]));
        float v1 = fmaxf(fmaxf(b1[0], b1[1]), fmaxf(b1[28], b1[29]));
        dst32[i] = (unsigned)f2bf(v0) | ((unsigned)f2bf(v1) << 16);
    }
}

// =====================================================================
// Kernel 4: FC1 (2048x12544 @ 12544x128^T) + bias + ReLU -> h4 bf16.
// =====================================================================
__global__ __launch_bounds__(256) void k_fc1(const unsigned short* __restrict__ h3,
                                             const float* __restrict__ fw1,
                                             const float* __restrict__ fb1,
                                             unsigned short* __restrict__ h4) {
    extern __shared__ char smem[];
    unsigned short* sA = (unsigned short*)smem;      // [128 rows][32 k]
    unsigned short* sB = sA + 128 * 32;              // [128 n][32 k]
    const int m0 = blockIdx.x * 128;
    const int wave = threadIdx.x >> 5, lane = threadIdx.x & 31;
    const int half = lane >> 4, lm = lane & 15;

    v8f acc[8];
    #pragma unroll
    for (int i = 0; i < 8; ++i) acc[i] = (v8f){};

    const unsigned* h3u = (const unsigned*)h3;

    for (int kk = 0; kk < 392; ++kk) {
        const int k0 = kk * 32;
        #pragma unroll
        for (int it = 0; it < 8; ++it) {
            int i = threadIdx.x + it * 256;          // 0..2047
            int row = i >> 4, jh = i & 15;
            int j = jh * 2;
            int e = j & 15;
            int hf = j >> 4;
            unsigned v = h3u[(((size_t)(m0 + row) * 12544) + k0 + j) >> 1];
            ((unsigned*)sA)[(row * 32 + hf * 16 + e) >> 1] = v;
        }
        #pragma unroll
        for (int it = 0; it < 8; ++it) {
            int i = threadIdx.x + it * 256;          // 0..2047
            int nc = i >> 4, jh = i & 15;
            const float2 f = *(const float2*)(fw1 + (size_t)nc * 12544 + k0 + jh * 2);
            ((unsigned*)sB)[(nc * 32 + jh * 2) >> 1] =
                (unsigned)f2bf(f.x) | ((unsigned)f2bf(f.y) << 16);
        }
        // prefetch next k-slab (global_prefetch_b8)
        if (kk + 1 < 392) {
            int row = threadIdx.x >> 1;
            __builtin_prefetch((const void*)(h3 + (size_t)(m0 + row) * 12544 + k0 + 32), 0, 1);
            __builtin_prefetch((const void*)(fw1 + (size_t)row * 12544 + k0 + 32), 0, 1);
        }
        __syncthreads();

        V16 ta; ta.u = *(const v16u*)(sA + (wave * 16 + lm) * 32 + half * 16);
        #pragma unroll
        for (int nt = 0; nt < 8; ++nt) {
            V16 tb; tb.u = *(const v16u*)(sB + (nt * 16 + lm) * 32 + half * 16);
            acc[nt] = __builtin_amdgcn_wmma_f32_16x16x32_bf16(false, ta.b, false, tb.b,
                                                              (short)0, acc[nt], false, false);
        }
        __syncthreads();
    }

    #pragma unroll
    for (int nt = 0; nt < 8; ++nt) {
        #pragma unroll
        for (int r = 0; r < 8; ++r) {
            int m = m0 + wave * 16 + r + half * 8;
            int nc = nt * 16 + lm;
            float v = acc[nt][r] + fb1[nc];
            h4[(size_t)m * 128 + nc] = f2bf(v > 0.f ? v : 0.f);
        }
    }
}

// =====================================================================
// Kernel 5: FC2 (128->10) + log_softmax. One thread per batch row.
// =====================================================================
__global__ __launch_bounds__(256) void k_fc2(const unsigned short* __restrict__ h4,
                                             const float* __restrict__ fw2,
                                             const float* __restrict__ fb2,
                                             float* __restrict__ out) {
    int row = blockIdx.x * 256 + threadIdx.x;
    if (row >= 2048) return;
    const unsigned* hu = (const unsigned*)(h4 + (size_t)row * 128);
    float logits[10];
    #pragma unroll
    for (int j = 0; j < 10; ++j) {
        float accv = fb2[j];
        const float* wr = fw2 + j * 128;
        for (int k = 0; k < 64; ++k) {
            unsigned w2v = hu[k];
            accv += bf2f((unsigned short)(w2v & 0xFFFFu)) * wr[2 * k];
            accv += bf2f((unsigned short)(w2v >> 16))     * wr[2 * k + 1];
        }
        logits[j] = accv;
    }
    float mx = logits[0];
    #pragma unroll
    for (int j = 1; j < 10; ++j) mx = fmaxf(mx, logits[j]);
    float se = 0.f;
    #pragma unroll
    for (int j = 0; j < 10; ++j) se += expf(logits[j] - mx);
    float lse = mx + logf(se);
    #pragma unroll
    for (int j = 0; j < 10; ++j) out[(size_t)row * 10 + j] = logits[j] - lse;
}

// =====================================================================
extern "C" void kernel_launch(void* const* d_in, const int* in_sizes, int n_in,
                              void* d_out, int out_size, void* d_ws, size_t ws_size,
                              hipStream_t stream) {
    (void)in_sizes; (void)n_in; (void)out_size; (void)ws_size;
    const float* x   = (const float*)d_in[0];
    const float* w1  = (const float*)d_in[1];
    const float* b1  = (const float*)d_in[2];
    const float* w2  = (const float*)d_in[3];
    const float* b2  = (const float*)d_in[4];
    const float* w3  = (const float*)d_in[5];
    const float* b3  = (const float*)d_in[6];
    const float* fw1 = (const float*)d_in[7];
    const float* fb1 = (const float*)d_in[8];
    const float* fw2 = (const float*)d_in[9];
    const float* fb2 = (const float*)d_in[10];

    char* ws = (char*)d_ws;
    unsigned short* h1  = (unsigned short*)(ws + 0);             // 2048*25088 bf16 [p][32]
    unsigned short* h2  = (unsigned short*)(ws + 102760448ull);  // 2048*50176 bf16 [p][64]
    unsigned short* wp2 = (unsigned short*)(ws + 308281344ull);  // 18432 bf16
    unsigned short* wp3 = (unsigned short*)(ws + 308318208ull);  // 36864 bf16
    unsigned short* h3  = (unsigned short*)(ws + 0);             // reuse: 2048*12544 bf16
    unsigned short* h4  = (unsigned short*)(ws + 51380224ull);   // 2048*128 bf16
    float* out = (float*)d_out;

    k_prep <<<32,   256, 0,      stream>>>(w2, w3, wp2, wp3);
    k_conv1<<<2048, 256, 105488, stream>>>(x, w1, b1, h1);
    k_conv2<<<2048, 256, 251456, stream>>>(h1, wp2, b2, h2);
    k_conv3<<<2048, 256, 301184, stream>>>(h2, wp3, b3, h3);
    k_fc1  <<<16,   256, 16384,  stream>>>(h3, fw1, fb1, h4);
    k_fc2  <<<8,    256, 0,      stream>>>(h4, fw2, fb2, out);
}
